// FlexOlmoMoE_8289286881926
// MI455X (gfx1250) — matrix operand order repaired and checked by reading, not compiled, not run on hardware
//
#include <hip/hip_runtime.h>
#include <hip/hip_bf16.h>
#include <math.h>

typedef __attribute__((ext_vector_type(16))) __bf16 v16bf;
typedef __attribute__((ext_vector_type(8)))  float  v8f;
typedef __attribute__((ext_vector_type(4)))  unsigned int u32x4;
typedef __attribute__((ext_vector_type(8)))  unsigned int u32x8;

static constexpr int T_TOK = 4096;   // tokens
static constexpr int DIM   = 2048;   // hidden
static constexpr int FDIM  = 4096;   // intermediate
static constexpr int NEXP  = 8;      // experts
static constexpr int LDK   = 40;     // padded K-stride in LDS (16B-aligned rows, bank spread)

// ---------------------------------------------------------------------------
// CDNA5 data-movement helpers
// ---------------------------------------------------------------------------
// Async copy 16B global -> LDS (bypasses VGPRs, tracked by ASYNCcnt).
__device__ __forceinline__ void async_g2lds_b128(void* lds, const void* g) {
  unsigned lds_off = (unsigned)(uintptr_t)lds;              // low 32b = LDS offset
  unsigned long long ga = (unsigned long long)(uintptr_t)g;
  asm volatile("global_load_async_to_lds_b128 %0, %1, off"
               :: "v"(lds_off), "v"(ga) : "memory");
}
__device__ __forceinline__ void wait_async0() {
  asm volatile("s_wait_asynccnt 0x0" ::: "memory");
}
__device__ __forceinline__ void wait_tensor0() {
  asm volatile("s_wait_tensorcnt 0x0" ::: "memory");
}
// TDM: DMA a 2D tile [rows x 32 bf16] from a row-major tensor into LDS with
// hardware padding producing row stride LDK(=40) elements (64B data + 16B pad).
__device__ __forceinline__ void tdm_load_tile_2d(void* lds, const void* g,
                                                 unsigned tensor_d0, unsigned tensor_d1,
                                                 unsigned stride_d0,
                                                 unsigned tile_d0, unsigned tile_d1) {
  unsigned long long ga = (unsigned long long)(uintptr_t)g;
  unsigned lds_off = (unsigned)(uintptr_t)lds;
  u32x4 g0;
  g0[0] = 1u;                                   // count=1, user descriptor
  g0[1] = lds_off;                              // lds_addr
  g0[2] = (unsigned)(ga & 0xFFFFFFFFu);         // global_addr[31:0]
  g0[3] = (unsigned)((ga >> 32) & 0x01FFFFFFu)  // global_addr[56:32]
          | (2u << 30);                         // type=2 (image)
  u32x8 g1;
  g1[0] = (1u << 16)      // data_size = 2 bytes
        | (1u << 20)      // pad_enable
        | (3u << 22)      // pad_interval: 16 DWORDs (=64B) between pads
        | (3u << 25);     // pad_amount: 4 DWORDs (=16B) -> 80B row stride
  g1[1] = (tensor_d0 & 0xFFFFu) << 16;                       // tensor_dim0[15:0]
  g1[2] = (tensor_d0 >> 16) | ((tensor_d1 & 0xFFFFu) << 16); // dim0 hi | dim1 lo
  g1[3] = (tensor_d1 >> 16) | (tile_d0 << 16);               // dim1 hi | tile_dim0
  g1[4] = tile_d1;                                           // tile_dim1 (tile_dim2=0)
  g1[5] = stride_d0;                                         // tensor_dim0_stride[31:0]
  g1[6] = 0u;                                                // stride hi | dim1_stride lo
  g1[7] = 0u;
  asm volatile("tensor_load_to_lds %0, %1" :: "s"(g0), "s"(g1) : "memory");
}

// ---------------------------------------------------------------------------
// WMMA helpers (gfx1250, wave32)
// ---------------------------------------------------------------------------
__device__ __forceinline__ v8f wmma_bf16(v16bf a, v16bf b, v8f c) {
  return __builtin_amdgcn_wmma_f32_16x16x32_bf16(false, a, false, b, (short)0, c,
                                                 false, false);
}

// A fragment (16x32 bf16, MxK) from LDS row-major [row][k], stride LDK.
// lane r<16: row r, K {0..7,16..23}; lane r+16: row r, K {8..15,24..31}.
__device__ __forceinline__ v16bf frag_a(const __bf16* tile, int rowBase) {
  int lane = threadIdx.x & 31;
  int r = lane & 15, h = lane >> 4;
  const __bf16* p = tile + (size_t)(rowBase + r) * LDK + h * 8;
  union { uint4 u[2]; v16bf v; } c;
  c.u[0] = *(const uint4*)(p);
  c.u[1] = *(const uint4*)(p + 16);
  return c.v;
}

// B fragment (32x16 bf16, KxN) from LDS stored TRANSPOSED [n][k], stride LDK.
// lane n<16: col n, K {0..7,16..23}; lane n+16: col n, K {8..15,24..31}.
__device__ __forceinline__ v16bf frag_b(const __bf16* tile, int colBase) {
  int lane = threadIdx.x & 31;
  int r = lane & 15, h = lane >> 4;
  const __bf16* p = tile + (size_t)(colBase + r) * LDK + h * 8;
  union { uint4 u[2]; v16bf v; } c;
  c.u[0] = *(const uint4*)(p);
  c.u[1] = *(const uint4*)(p + 16);
  return c.v;
}

// ---------------------------------------------------------------------------
// Kernel 0: fp32 -> bf16 conversion of activations
// ---------------------------------------------------------------------------
__global__ __launch_bounds__(256) void cvt_x_kernel(const float* __restrict__ x,
                                                    __bf16* __restrict__ xb, int n) {
  int i = (blockIdx.x * blockDim.x + threadIdx.x) * 8;
  if (i >= n) return;
  float4 a = *(const float4*)(x + i);
  float4 b = *(const float4*)(x + i + 4);
  union { __bf16 h[8]; uint4 u; } o;
  o.h[0] = (__bf16)a.x; o.h[1] = (__bf16)a.y; o.h[2] = (__bf16)a.z; o.h[3] = (__bf16)a.w;
  o.h[4] = (__bf16)b.x; o.h[5] = (__bf16)b.y; o.h[6] = (__bf16)b.z; o.h[7] = (__bf16)b.w;
  *(uint4*)(xb + i) = o.u;
}

// ---------------------------------------------------------------------------
// Kernel 1: router — logits, softmax over all experts, top-2 (no renorm)
// ---------------------------------------------------------------------------
__global__ __launch_bounds__(256) void router_kernel(const float* __restrict__ x,
                                                     const float* __restrict__ gw,
                                                     float* __restrict__ combine) {
  int wave = threadIdx.x >> 5;
  int lane = threadIdx.x & 31;
  int t = blockIdx.x * (blockDim.x >> 5) + wave;
  if (t >= T_TOK) return;

  float acc[NEXP];
#pragma unroll
  for (int e = 0; e < NEXP; e++) acc[e] = 0.f;

  const float* xr = x + (size_t)t * DIM;
  for (int d = lane; d < DIM; d += 32) {
    float xv = xr[d];
    const float* g = gw + (size_t)d * NEXP;
#pragma unroll
    for (int e = 0; e < NEXP; e++) acc[e] += xv * g[e];
  }
#pragma unroll
  for (int e = 0; e < NEXP; e++) {
#pragma unroll
    for (int off = 16; off > 0; off >>= 1) acc[e] += __shfl_xor(acc[e], off, 32);
  }
  if (lane == 0) {
    float m = acc[0];
#pragma unroll
    for (int e = 1; e < NEXP; e++) m = fmaxf(m, acc[e]);
    float p[NEXP], s = 0.f;
#pragma unroll
    for (int e = 0; e < NEXP; e++) { p[e] = __expf(acc[e] - m); s += p[e]; }
    float inv = __builtin_amdgcn_rcpf(s);
#pragma unroll
    for (int e = 0; e < NEXP; e++) p[e] *= inv;
    int i1 = 0;
#pragma unroll
    for (int e = 1; e < NEXP; e++) if (p[e] > p[i1]) i1 = e;
    int i2 = (i1 == 0) ? 1 : 0;
#pragma unroll
    for (int e = 0; e < NEXP; e++) if (e != i1 && p[e] > p[i2]) i2 = e;
    float* c = combine + (size_t)t * NEXP;
#pragma unroll
    for (int e = 0; e < NEXP; e++) c[e] = (e == i1 || e == i2) ? p[e] : 0.f;
  }
}

// ---------------------------------------------------------------------------
// Kernel 2: per-expert SwiGLU: mid[e] = combine[:,e] * silu(xb@w1e) * (xb@w3e)
// Tile 128 tok x 64 F. A-tile staged via global_load_async_to_lds_b128.
// ---------------------------------------------------------------------------
__global__ __launch_bounds__(256) void expert_mid_kernel(
    const __bf16* __restrict__ xb, const float* __restrict__ w1,
    const float* __restrict__ w3, const float* __restrict__ combine,
    __bf16* __restrict__ mid) {
  __shared__ __bf16 sA[128 * LDK];
  __shared__ __bf16 sB1[64 * LDK];
  __shared__ __bf16 sB3[64 * LDK];
  __shared__ float  sC[128];

  const int e  = blockIdx.z;
  const int t0 = blockIdx.y * 128;
  const int f0 = blockIdx.x * 64;
  const int tid = threadIdx.x;
  const int lane = tid & 31, wave = tid >> 5;
  const int wm = wave & 3, wn = wave >> 2;   // 4 x 2 waves

  const float* w1e = w1 + (size_t)e * DIM * FDIM;
  const float* w3e = w3 + (size_t)e * DIM * FDIM;

  if (tid < 128) sC[tid] = combine[(size_t)(t0 + tid) * NEXP + e];

  v8f accg[2][2] = {};
  v8f accu[2][2] = {};

  for (int k0 = 0; k0 < DIM; k0 += 32) {
    __syncthreads();
    // --- stage A via async global->LDS (verbatim bf16, no conversion) ---
    {
      int row = tid >> 1, part = (tid & 1) * 16;
      const __bf16* g = xb + (size_t)(t0 + row) * DIM + k0 + part;
      async_g2lds_b128(&sA[row * LDK + part],     g);
      async_g2lds_b128(&sA[row * LDK + part + 8], g + 8);
    }
    // --- stage B1/B3: [32 d x 64 f] fp32 -> bf16, transposed to [f][d] ---
    {
      int dr = tid >> 3, fc = (tid & 7) * 8;
      const float* g1 = w1e + (size_t)(k0 + dr) * FDIM + f0 + fc;
      const float* g3 = w3e + (size_t)(k0 + dr) * FDIM + f0 + fc;
      float4 a0 = *(const float4*)(g1);
      float4 a1 = *(const float4*)(g1 + 4);
      float4 b0 = *(const float4*)(g3);
      float4 b1 = *(const float4*)(g3 + 4);
      float v1[8] = {a0.x, a0.y, a0.z, a0.w, a1.x, a1.y, a1.z, a1.w};
      float v3[8] = {b0.x, b0.y, b0.z, b0.w, b1.x, b1.y, b1.z, b1.w};
#pragma unroll
      for (int j = 0; j < 8; j++) {
        sB1[(size_t)(fc + j) * LDK + dr] = (__bf16)v1[j];
        sB3[(size_t)(fc + j) * LDK + dr] = (__bf16)v3[j];
      }
    }
    wait_async0();
    __syncthreads();
    // --- compute: 8 WMMA per K-step ---
    v16bf a0 = frag_a(sA, wm * 32);
    v16bf a1 = frag_a(sA, wm * 32 + 16);
#pragma unroll
    for (int bn = 0; bn < 2; bn++) {
      v16bf b1f = frag_b(sB1, wn * 32 + bn * 16);
      v16bf b3f = frag_b(sB3, wn * 32 + bn * 16);
      accg[0][bn] = wmma_bf16(a0, b1f, accg[0][bn]);
      accg[1][bn] = wmma_bf16(a1, b1f, accg[1][bn]);
      accu[0][bn] = wmma_bf16(a0, b3f, accu[0][bn]);
      accu[1][bn] = wmma_bf16(a1, b3f, accu[1][bn]);
    }
  }

  // --- epilogue: silu(g)*u, scaled by combine weight, store bf16 ---
  int r = lane & 15, h = lane >> 4;
#pragma unroll
  for (int am = 0; am < 2; am++) {
#pragma unroll
    for (int bn = 0; bn < 2; bn++) {
      v8f g = accg[am][bn];
      v8f u = accu[am][bn];
      int col = f0 + wn * 32 + bn * 16 + r;
#pragma unroll
      for (int i = 0; i < 8; i++) {
        int ridx = wm * 32 + am * 16 + h * 8 + i;
        float gv = g[i];
        float sval = gv * __builtin_amdgcn_rcpf(1.f + __expf(-gv));  // silu
        float o = sC[ridx] * sval * u[i];
        mid[((size_t)e * T_TOK + (t0 + ridx)) * FDIM + col] = (__bf16)o;
      }
    }
  }
}

// ---------------------------------------------------------------------------
// Kernel 3: down projection accumulated over all experts (mid pre-scaled):
//   out[t,d] = sum_e sum_f mid[e,t,f] * w2[e,f,d]
// Tile 128 tok x 128 D. A-tile staged via TDM tensor_load_to_lds (wave 0).
// ---------------------------------------------------------------------------
__global__ __launch_bounds__(256) void down_proj_kernel(
    const __bf16* __restrict__ mid, const float* __restrict__ w2,
    float* __restrict__ out) {
  __shared__ __bf16 sA[128 * LDK];
  __shared__ __bf16 sB[128 * LDK];

  const int t0 = blockIdx.y * 128;
  const int d0 = blockIdx.x * 128;
  const int tid = threadIdx.x, lane = tid & 31, wave = tid >> 5;
  const int wm = wave & 3, wn = wave >> 2;   // 4 x 2 waves; wave tile 32 x 64

  v8f acc[2][4] = {};

  for (int e = 0; e < NEXP; e++) {
    const __bf16* me  = mid + (size_t)e * T_TOK * FDIM;
    const float*  w2e = w2  + (size_t)e * FDIM * DIM;
    for (int k0 = 0; k0 < FDIM; k0 += 32) {
      __syncthreads();
      // --- stage A: TDM DMA of [128 rows x 32 k] bf16 tile (wave 0 only),
      //     hardware pad -> LDS row stride = LDK elements ---
      if (wave == 0) {
        const __bf16* g = me + (size_t)t0 * FDIM + k0;
        tdm_load_tile_2d(sA, g, /*tensor_d0=*/FDIM, /*tensor_d1=*/T_TOK,
                         /*stride_d0=*/FDIM, /*tile_d0=*/32, /*tile_d1=*/128);
      }
      // --- stage B: w2 tile [32 f x 128 d] fp32 -> bf16, transposed ---
      {
        int fr = tid >> 3, dc = (tid & 7) * 16;
        const float* gw = w2e + (size_t)(k0 + fr) * DIM + d0 + dc;
        if (k0 + 32 < FDIM)
          __builtin_prefetch(gw + (size_t)32 * DIM, 0, 1);
        float4 c0 = *(const float4*)(gw);
        float4 c1 = *(const float4*)(gw + 4);
        float4 c2 = *(const float4*)(gw + 8);
        float4 c3 = *(const float4*)(gw + 12);
        float v[16] = {c0.x, c0.y, c0.z, c0.w, c1.x, c1.y, c1.z, c1.w,
                       c2.x, c2.y, c2.z, c2.w, c3.x, c3.y, c3.z, c3.w};
#pragma unroll
        for (int j = 0; j < 16; j++)
          sB[(size_t)(dc + j) * LDK + fr] = (__bf16)v[j];
      }
      if (wave == 0) wait_tensor0();
      __syncthreads();
      // --- compute: 8 WMMA per K-step ---
      v16bf a0 = frag_a(sA, wm * 32);
      v16bf a1 = frag_a(sA, wm * 32 + 16);
#pragma unroll
      for (int bn = 0; bn < 4; bn++) {
        v16bf b = frag_b(sB, wn * 64 + bn * 16);
        acc[0][bn] = wmma_bf16(a0, b, acc[0][bn]);
        acc[1][bn] = wmma_bf16(a1, b, acc[1][bn]);
      }
    }
  }

  // --- epilogue: write fp32 output ---
  int r = lane & 15, h = lane >> 4;
#pragma unroll
  for (int am = 0; am < 2; am++) {
#pragma unroll
    for (int bn = 0; bn < 4; bn++) {
      int col = d0 + wn * 64 + bn * 16 + r;
#pragma unroll
      for (int i = 0; i < 8; i++) {
        int row = t0 + wm * 32 + am * 16 + h * 8 + i;
        out[(size_t)row * DIM + col] = acc[am][bn][i];
      }
    }
  }
}

// ---------------------------------------------------------------------------
// Launch
// ---------------------------------------------------------------------------
extern "C" void kernel_launch(void* const* d_in, const int* in_sizes, int n_in,
                              void* d_out, int out_size, void* d_ws, size_t ws_size,
                              hipStream_t stream) {
  const float* x  = (const float*)d_in[0];   // [T, D]
  const float* gw = (const float*)d_in[1];   // [D, E]
  const float* w1 = (const float*)d_in[2];   // [E, D, F]
  const float* w3 = (const float*)d_in[3];   // [E, D, F]
  const float* w2 = (const float*)d_in[4];   // [E, F, D]
  float* out = (float*)d_out;                // [T, D]

  char* ws = (char*)d_ws;
  __bf16* xb      = (__bf16*)ws;                                     // 16 MiB
  float*  combine = (float*)(ws + (size_t)T_TOK * DIM * 2);          // 128 KiB
  __bf16* mid     = (__bf16*)(ws + (size_t)T_TOK * DIM * 2
                                 + (size_t)T_TOK * NEXP * 4);        // 256 MiB

  {
    int n = T_TOK * DIM;
    cvt_x_kernel<<<n / (256 * 8), 256, 0, stream>>>(x, xb, n);
  }
  router_kernel<<<T_TOK / 8, 256, 0, stream>>>(x, gw, combine);
  {
    dim3 grid(FDIM / 64, T_TOK / 128, NEXP);
    expert_mid_kernel<<<grid, 256, 0, stream>>>(xb, w1, w3, combine, mid);
  }
  {
    dim3 grid(DIM / 128, T_TOK / 128);
    down_proj_kernel<<<grid, 256, 0, stream>>>(mid, w2, out);
  }
}
